// MutiheadAttention_55216099558001
// MI455X (gfx1250) — matrix operand order repaired and checked by reading, not compiled, not run on hardware
//
#include <hip/hip_runtime.h>
#include <hip/hip_bf16.h>

typedef _Float16 h16;
typedef __attribute__((ext_vector_type(16))) _Float16 v16h;
typedef __attribute__((ext_vector_type(8)))  _Float16 v8h;
typedef __attribute__((ext_vector_type(8)))  float    v8f;
typedef __attribute__((ext_vector_type(4)))  int      v4i;

#define B_    8
#define T_    512
#define F_    512
#define H_    8
#define DH    64
#define HB_   64
#define ML    2048
#define SPAST 1536   // ML - T
#define EPS_  1e-8f

#define WMMA_F16(a, b, c) \
  __builtin_amdgcn_wmma_f32_16x16x32_f16(false, (a), false, (b), (short)0, (c), false, false)

// ---------------- CDNA5 async global->LDS copy (ASYNCcnt path) ----------------

__device__ __forceinline__ void async_copy_b128(const h16* gsrc, h16* ldst) {
#if __has_builtin(__builtin_amdgcn_global_load_async_to_lds_b128)
  h16* gs = const_cast<h16*>(gsrc);
  __builtin_amdgcn_global_load_async_to_lds_b128(
      (__attribute__((address_space(1))) v4i*)gs,
      (__attribute__((address_space(3))) v4i*)ldst, 0, 0);
#else
  unsigned loff = (unsigned)(unsigned long long)
      (__attribute__((address_space(3))) h16*)ldst;
  asm volatile("global_load_async_to_lds_b128 %0, %1, off"
               :: "v"(loff), "v"(gsrc) : "memory");
#endif
}

__device__ __forceinline__ void wait_async0() {
#if __has_builtin(__builtin_amdgcn_s_wait_asynccnt)
  __builtin_amdgcn_s_wait_asynccnt(0);
#else
  asm volatile("s_wait_asynccnt 0" ::: "memory");
#endif
}

// ---------------- fragment loaders (wave32, 16-bit WMMA layouts) ----------------

// A-matrix 16x32 (MxK), f16. base points at row m0, row_stride in halfs.
// lanes 0-15: M=lane, K = k0+[0..15]
// lanes 16-31: M=lane-16, K = k0+[16..31]
__device__ __forceinline__ v16h load_a_frag(const h16* base, int ln,
                                            int row_stride, int k0) {
  int kb = k0 + ((ln & 16) ? 16 : 0);
  const h16* p = base + (ln & 15) * row_stride + kb;
  v8h lo = *(const v8h*)(p);
  v8h hi = *(const v8h*)(p + 8);
  v16h a;
#pragma unroll
  for (int i = 0; i < 8; ++i) { a[i] = lo[i]; a[i + 8] = hi[i]; }
  return a;
}

// B-matrix 32x16 (KxN), f16, stored in memory as [N][K] row-major (i.e. B^T).
// lane holds column N = lane%16, K = k0 + (lane>=16?16:0) .. +15 (contiguous 32B).
__device__ __forceinline__ v16h load_b_frag(const h16* base, int ln,
                                            int row_stride, int k0) {
  int kb = k0 + ((ln & 16) ? 16 : 0);
  return *(const v16h*)(base + (ln & 15) * row_stride + kb);
}

// ---------------- prep kernels ----------------

__global__ void cvt_f32_to_f16(const float* __restrict__ src, h16* __restrict__ dst, int n) {
  int i = blockIdx.x * blockDim.x + threadIdx.x;
  if (i < n) dst[i] = (h16)src[i];
}

// Rolling window: Kf[hb][s][d] = cache_k[hb][T+s][d], Vt[hb][d][s] = cache_v[hb][T+s][d]
__global__ void prep_kv(const float* __restrict__ ck, const float* __restrict__ cv,
                        h16* __restrict__ Kf, h16* __restrict__ Vt) {
  int i = blockIdx.x * blockDim.x + threadIdx.x;
  const int total = HB_ * SPAST * DH;
  if (i >= total) return;
  int d  = i % DH;
  int s  = (i / DH) % SPAST;
  int hb = i / (DH * SPAST);
  int src = (hb * ML + (T_ + s)) * DH + d;
  Kf[(hb * ML + s) * DH + d] = (h16)ck[src];
  Vt[(hb * DH + d) * ML + s] = (h16)cv[src];
}

// ---------------- QKV projection: C = X @ W^T + b, scattered to head layouts ----------------

__global__ __launch_bounds__(256) void qkv_gemm(const h16* __restrict__ Xh,
                                                const h16* __restrict__ Wh3,  // [3][F][F]
                                                const float* __restrict__ bq,
                                                const float* __restrict__ bk,
                                                const float* __restrict__ bv,
                                                h16* __restrict__ Qh,
                                                h16* __restrict__ Kf,
                                                h16* __restrict__ Vt) {
  const int which = blockIdx.y;
  const h16* W = Wh3 + which * F_ * F_;
  const float* bias = (which == 0) ? bq : (which == 1) ? bk : bv;

  int ln   = threadIdx.x & 31;
  int tile = blockIdx.x * 8 + (threadIdx.x >> 5);  // 2048 wave-tiles
  int m0   = (tile >> 3) * 16;                     // 256 M-tiles
  int n0   = (tile & 7) * 64;                      // 8 N-tiles of 64

  v8f acc[4] = {};
  for (int k0 = 0; k0 < F_; k0 += 32) {
    __builtin_prefetch(Xh + m0 * F_ + k0 + 64, 0, 0);  // gfx1250 global_prefetch_b8
    v16h a = load_a_frag(Xh + m0 * F_, ln, F_, k0);
#pragma unroll
    for (int q = 0; q < 4; ++q) {
      v16h b = load_b_frag(W + (n0 + q * 16) * F_, ln, F_, k0);
      acc[q] = WMMA_F16(a, b, acc[q]);
    }
  }

  int rlo = ln & 15;
  int rh  = (ln & 16) ? 8 : 0;
#pragma unroll
  for (int q = 0; q < 4; ++q) {
#pragma unroll
    for (int j = 0; j < 8; ++j) {
      int m = m0 + j + rh;
      int n = n0 + q * 16 + rlo;
      float val = acc[q][j] + bias[n];
      int bb = m / T_, t = m % T_;
      int h = n / DH, dc = n % DH;
      int hb = h * B_ + bb;  // head index outermost
      h16 hv = (h16)val;
      if (which == 0)      Qh[(hb * T_ + t) * DH + dc] = hv;
      else if (which == 1) Kf[(hb * ML + (SPAST + t)) * DH + dc] = hv;
      else                 Vt[(hb * DH + dc) * ML + (SPAST + t)] = hv;
    }
  }
}

// ---------------- fused attention with async double-buffered K/V staging ----------------
// All 8 waves of a block share one head-batch hb; wave 0 streams the shared
// 32(s)x64(d) K tile and 64(d)x32(s) V tile into LDS with async-to-LDS copies.

__global__ __launch_bounds__(256) void attn_kernel(const h16* __restrict__ Qh,
                                                   const h16* __restrict__ Kf,
                                                   const h16* __restrict__ Vt,
                                                   const float* __restrict__ delta,
                                                   h16* __restrict__ Om) {
  __shared__ __align__(64) h16 KT[2][32 * DH];   // [buf][s][d]   2 x 4KB
  __shared__ __align__(64) h16 VT[2][DH * 32];   // [buf][d][s]   2 x 4KB
  __shared__ __align__(64) h16 PT[8][16 * 32];   // per-wave P    8 x 1KB

  int ln = threadIdx.x & 31;
  int wv = threadIdx.x >> 5;
  int tile = blockIdx.x * 8 + wv;        // 2048 tiles = 64 hb * 32 t-tiles
  int hb = blockIdx.x >> 2;              // uniform across the block
  int t0 = (tile & 31) << 4;

  const h16* Kg = Kf + (size_t)hb * ML * DH;  // [s][d], tiles are contiguous 4KB
  const h16* Vg = Vt + (size_t)hb * DH * ML;  // [d][ML]

  float dv = delta[0];
  float invDen = 1.0f / (dv * dv + EPS_);
  const float invD = 0.044194173824159216f;  // 1/sqrt(512)

  v16h qa0 = load_a_frag(Qh + (hb * T_ + t0) * DH, ln, DH, 0);
  v16h qa1 = load_a_frag(Qh + (hb * T_ + t0) * DH, ln, DH, 32);

  v8f oacc[4] = {};
  float mrow[8], lrow[8];
#pragma unroll
  for (int j = 0; j < 8; ++j) { mrow[j] = 0.0f; lrow[j] = 0.0f; }

  int rlo = ln & 15;
  int rh  = (ln & 16) ? 8 : 0;
  h16* pl = PT[wv];

  // prologue: stage chunk 0 into buffer 0
  if (wv == 0) {
#pragma unroll
    for (int it = 0; it < 8; ++it) {
      int u = it * 32 + ln;                       // 16B unit index 0..255
      async_copy_b128(Kg + u * 8, &KT[0][u * 8]); // K tile is contiguous
      int d = u >> 2, sub = u & 3;
      async_copy_b128(Vg + (size_t)d * ML + sub * 8, &VT[0][d * 32 + sub * 8]);
    }
    wait_async0();
  }
  __syncthreads();

  for (int s0 = 0; s0 < ML; s0 += 32) {
    int cur = (s0 >> 5) & 1;
    int nxt = cur ^ 1;

    // kick off async staging of the next chunk while computing on this one
    if (wv == 0 && (s0 + 32) < ML) {
      const h16* Kg2 = Kg + (s0 + 32) * DH;
#pragma unroll
      for (int it = 0; it < 8; ++it) {
        int u = it * 32 + ln;
        async_copy_b128(Kg2 + u * 8, &KT[nxt][u * 8]);
        int d = u >> 2, sub = u & 3;
        async_copy_b128(Vg + (size_t)d * ML + (s0 + 32) + sub * 8,
                        &VT[nxt][d * 32 + sub * 8]);
      }
    }

    // S tile: 16(t) x 32(s), two 16x16 C-tiles, K over dh=64 in two WMMA steps
    const h16* kb0 = &KT[cur][0];
    const h16* kb1 = &KT[cur][16 * DH];
    v8f sc0 = {}, sc1 = {};
    sc0 = WMMA_F16(qa0, load_b_frag(kb0, ln, DH, 0),  sc0);
    sc0 = WMMA_F16(qa1, load_b_frag(kb0, ln, DH, 32), sc0);
    sc1 = WMMA_F16(qa0, load_b_frag(kb1, ln, DH, 0),  sc1);
    sc1 = WMMA_F16(qa1, load_b_frag(kb1, ln, DH, 32), sc1);

    // Gaussian bias + |.|/sqrt(F) + online softmax (row j+(rh), col rlo)
    float p0[8], p1[8];
#pragma unroll
    for (int j = 0; j < 8; ++j) {
      float ti = (float)(SPAST + t0 + j + rh);
      float s_a = (float)(s0 + rlo);
      float s_b = (float)(s0 + 16 + rlo);
      float ga = ti - s_a, gb = ti - s_b;
      float va = fabsf(sc0[j] * __expf(-ga * ga * invDen) * invD);
      float vb = fabsf(sc1[j] * __expf(-gb * gb * invDen) * invD);
      float vmax = fmaxf(va, vb);
#pragma unroll
      for (int off = 8; off; off >>= 1)
        vmax = fmaxf(vmax, __shfl_xor(vmax, off, 32));
      float newm = fmaxf(mrow[j], vmax);
      float scale = __expf(mrow[j] - newm);
      mrow[j] = newm;
      float ea = __expf(va - newm);
      float eb = __expf(vb - newm);
      p0[j] = ea; p1[j] = eb;
      float rsum = ea + eb;
#pragma unroll
      for (int off = 8; off; off >>= 1)
        rsum += __shfl_xor(rsum, off, 32);
      lrow[j] = lrow[j] * scale + rsum;
#pragma unroll
      for (int dt = 0; dt < 4; ++dt) oacc[dt][j] *= scale;
    }

    // Stage P (C-layout f32 -> row-major f16 tile in LDS)
#pragma unroll
    for (int j = 0; j < 8; ++j) {
      pl[(j + rh) * 32 + rlo]      = (h16)p0[j];
      pl[(j + rh) * 32 + 16 + rlo] = (h16)p1[j];
    }
    __syncthreads();
    v16h ap = load_a_frag(pl, ln, 32, 0);

    // O += P @ V  (B from LDS VT[d][s_local]: [N=d][K=s] row-major, stride 32)
#pragma unroll
    for (int dt = 0; dt < 4; ++dt) {
      v16h vb = load_b_frag(&VT[cur][dt * 16 * 32], ln, 32, 0);
      oacc[dt] = WMMA_F16(ap, vb, oacc[dt]);
    }

    // issuer drains its async queue, then the block flips buffers
    if (wv == 0) wait_async0();
    __syncthreads();
  }

  // finalize: divide by row sum, merge heads into Om[b][t][h*64+d] (f16)
  int h = hb / B_, bb = hb % B_;
#pragma unroll
  for (int dt = 0; dt < 4; ++dt) {
#pragma unroll
    for (int j = 0; j < 8; ++j) {
      int t = t0 + j + rh;
      int d = dt * 16 + rlo;
      float o = oacc[dt][j] / lrow[j];
      Om[(bb * T_ + t) * F_ + h * DH + d] = (h16)o;
    }
  }
}

// ---------------- output projection: out = Om @ Wo^T + bo (f32 out) ----------------

__global__ __launch_bounds__(256) void out_gemm(const h16* __restrict__ Ah,
                                                const h16* __restrict__ Who,
                                                const float* __restrict__ bias,
                                                float* __restrict__ out) {
  int ln   = threadIdx.x & 31;
  int tile = blockIdx.x * 8 + (threadIdx.x >> 5);
  int m0   = (tile >> 3) * 16;
  int n0   = (tile & 7) * 64;

  v8f acc[4] = {};
  for (int k0 = 0; k0 < F_; k0 += 32) {
    __builtin_prefetch(Ah + m0 * F_ + k0 + 64, 0, 0);
    v16h a = load_a_frag(Ah + m0 * F_, ln, F_, k0);
#pragma unroll
    for (int q = 0; q < 4; ++q) {
      v16h b = load_b_frag(Who + (n0 + q * 16) * F_, ln, F_, k0);
      acc[q] = WMMA_F16(a, b, acc[q]);
    }
  }

  int rlo = ln & 15;
  int rh  = (ln & 16) ? 8 : 0;
#pragma unroll
  for (int q = 0; q < 4; ++q) {
#pragma unroll
    for (int j = 0; j < 8; ++j) {
      int m = m0 + j + rh;
      int n = n0 + q * 16 + rlo;
      out[m * F_ + n] = acc[q][j] + bias[n];
    }
  }
}

// ---------------- host launcher ----------------

extern "C" void kernel_launch(void* const* d_in, const int* in_sizes, int n_in,
                              void* d_out, int out_size, void* d_ws, size_t ws_size,
                              hipStream_t stream) {
  (void)in_sizes; (void)n_in; (void)out_size; (void)ws_size;
  const float* x     = (const float*)d_in[0];
  const float* Wq    = (const float*)d_in[1];
  const float* bq    = (const float*)d_in[2];
  const float* Wk    = (const float*)d_in[3];
  const float* bk    = (const float*)d_in[4];
  const float* Wv    = (const float*)d_in[5];
  const float* bv    = (const float*)d_in[6];
  const float* Wo    = (const float*)d_in[7];
  const float* bo    = (const float*)d_in[8];
  const float* delta = (const float*)d_in[9];
  const float* ck    = (const float*)d_in[10];
  const float* cv    = (const float*)d_in[11];
  float* out = (float*)d_out;

  // workspace carve (all f16 buffers)
  char* ws = (char*)d_ws;
  h16* Xh  = (h16*)ws;  ws += (size_t)(B_ * T_) * F_ * 2;       // 4 MiB
  h16* Wh3 = (h16*)ws;  ws += (size_t)3 * F_ * F_ * 2;          // 1.5 MiB (Wq,Wk,Wv)
  h16* Who = (h16*)ws;  ws += (size_t)F_ * F_ * 2;              // 0.5 MiB
  h16* Qh  = (h16*)ws;  ws += (size_t)HB_ * T_ * DH * 2;        // 4 MiB
  h16* Kf  = (h16*)ws;  ws += (size_t)HB_ * ML * DH * 2;        // 16 MiB
  h16* Vt  = (h16*)ws;  ws += (size_t)HB_ * DH * ML * 2;        // 16 MiB
  h16* Om  = (h16*)ws;  ws += (size_t)(B_ * T_) * F_ * 2;       // 4 MiB

  const int XN = B_ * T_ * F_;
  const int WN = F_ * F_;

  cvt_f32_to_f16<<<XN / 256, 256, 0, stream>>>(x, Xh, XN);
  cvt_f32_to_f16<<<WN / 256, 256, 0, stream>>>(Wq, Wh3,          WN);
  cvt_f32_to_f16<<<WN / 256, 256, 0, stream>>>(Wk, Wh3 + WN,     WN);
  cvt_f32_to_f16<<<WN / 256, 256, 0, stream>>>(Wv, Wh3 + 2 * WN, WN);
  cvt_f32_to_f16<<<WN / 256, 256, 0, stream>>>(Wo, Who,          WN);

  const int KVN = HB_ * SPAST * DH;
  prep_kv<<<KVN / 256, 256, 0, stream>>>(ck, cv, Kf, Vt);

  // 2048 wave-tiles (16x64 each) per projection, 8 waves per block
  qkv_gemm<<<dim3(256, 3), 256, 0, stream>>>(Xh, Wh3, bq, bk, bv, Qh, Kf, Vt);

  // 64 head-batches x 32 t-tiles = 2048 wave-tiles, 8 waves per block
  attn_kernel<<<256, 256, 0, stream>>>(Qh, Kf, Vt, delta, Om);

  out_gemm<<<256, 256, 0, stream>>>(Om, Who, bo, out);
}
